// Model_39676907881153
// MI455X (gfx1250) — matrix operand order repaired and checked by reading, not compiled, not run on hardware
//
#include <hip/hip_runtime.h>

typedef __attribute__((ext_vector_type(16))) _Float16 v16h;
typedef __attribute__((ext_vector_type(8)))  _Float16 v8h;
typedef __attribute__((ext_vector_type(8)))  float    v8f;

#define B_ 2
#define S_ 2048
#define D_ 64
#define H_ 8
#define TK 64                      // keys per chunk
#define NCHUNK (S_ / TK)           // 32
#define ROWS_PER_WAVE 16
#define WAVES 8
#define ROWS_PER_BLOCK (ROWS_PER_WAVE * WAVES)   // 128

// ds_swizzle group-of-32 XOR butterfly max (offset = and 0x1F | xor<<10)
#define SWZ_MAX(x, imm)                                                        \
    x = fmaxf(x, __builtin_bit_cast(float, __builtin_amdgcn_ds_swizzle(       \
                     __builtin_bit_cast(int, x), (imm))))

__global__ __launch_bounds__(256)
void attn_bias_flash_kernel(const float* __restrict__ x1,   // Q [B,S,D]
                            const float* __restrict__ x2,   // K [B,S,D]
                            const float* __restrict__ x3,   // bias [B,H,S,S]
                            const float* __restrict__ x4,   // V [B,S,D]
                            float* __restrict__ out)        // [B,H,S,D]
{
    __shared__ _Float16 ldsK[2][D_][TK];              // [buf][d][key]  16 KB
    __shared__ _Float16 ldsV[2][TK][D_];              // [buf][key][d]  16 KB
    __shared__ _Float16 ldsP[WAVES][2][2][16][16];    // A-layout P     16 KB

    const int tid  = threadIdx.x;
    const int lane = tid & 31;
    const int wave = tid >> 5;
    const int half = lane >> 4;
    const int l15  = lane & 15;

    const int RB  = S_ / ROWS_PER_BLOCK;              // 16
    const int bid = blockIdx.x;
    const int b   = bid / (H_ * RB);
    const int h   = (bid / RB) % H_;
    const int rb  = bid % RB;
    const int row0 = rb * ROWS_PER_BLOCK + wave * ROWS_PER_WAVE;

    // staging assignment: 4 threads per key row, 16 d-values each (64B reads)
    const int skey = tid >> 2;
    const int sdg  = (tid & 3) * 16;

    const float* ksrc = x2 + (size_t)b * S_ * D_;
    const float* vsrc = x4 + (size_t)b * S_ * D_;
    const float* bias = x3 + (((size_t)b * H_ + h) * S_ + row0) * S_;

    // ---- Q A-fragments (1/8 scale folded in; exact since power of two) ----
    // A 16x32 f16 layout: m = lane&15; elem e: K=(v&3)*2+(v>>2)*16+half*8+(e&1)
    v16h qa[2];
    const float* qbase = x1 + ((size_t)b * S_ + row0 + l15) * D_;
    #pragma unroll
    for (int f = 0; f < 2; ++f) {
        #pragma unroll
        for (int e = 0; e < 16; ++e) {
            int v = e >> 1;
            int d = f * 32 + (v & 3) * 2 + (v >> 2) * 16 + half * 8 + (e & 1);
            qa[f][e] = (_Float16)(qbase[d] * 0.125f);
        }
    }

    v16h ones;
    #pragma unroll
    for (int e = 0; e < 16; ++e) ones[e] = (_Float16)1.0f;

    v8f acc[4] = {};                 // O: 4 d-tiles (row = r + 8*half)
    float m_run[8], l_run[8];
    #pragma unroll
    for (int r = 0; r < 8; ++r) { m_run[r] = -1e30f; l_run[r] = 0.f; }

    // ---- prologue: stage chunk 0 into buffer 0 ----
    {
        const float4* kp = (const float4*)(ksrc + (size_t)skey * D_ + sdg);
        const float4* vp = (const float4*)(vsrc + (size_t)skey * D_ + sdg);
        float4 kr[4] = {kp[0], kp[1], kp[2], kp[3]};
        float4 vr[4] = {vp[0], vp[1], vp[2], vp[3]};
        #pragma unroll
        for (int q = 0; q < 4; ++q) {
            ldsK[0][sdg + q * 4 + 0][skey] = (_Float16)kr[q].x;
            ldsK[0][sdg + q * 4 + 1][skey] = (_Float16)kr[q].y;
            ldsK[0][sdg + q * 4 + 2][skey] = (_Float16)kr[q].z;
            ldsK[0][sdg + q * 4 + 3][skey] = (_Float16)kr[q].w;
        }
        v8h lo, hi;
        #pragma unroll
        for (int q = 0; q < 2; ++q) {
            lo[q*4+0]=(_Float16)vr[q].x;   lo[q*4+1]=(_Float16)vr[q].y;
            lo[q*4+2]=(_Float16)vr[q].z;   lo[q*4+3]=(_Float16)vr[q].w;
            hi[q*4+0]=(_Float16)vr[q+2].x; hi[q*4+1]=(_Float16)vr[q+2].y;
            hi[q*4+2]=(_Float16)vr[q+2].z; hi[q*4+3]=(_Float16)vr[q+2].w;
        }
        *(v8h*)&ldsV[0][skey][sdg]     = lo;
        *(v8h*)&ldsV[0][skey][sdg + 8] = hi;
    }
    __syncthreads();

    for (int i = 0; i < NCHUNK; ++i) {
        const int kb  = i * TK;
        const int cur = i & 1;
        const bool have_next = (i + 1 < NCHUNK);

        // ---- issue next chunk's global loads FIRST (hidden under WMMA) ----
        float4 kr[4], vr[4];
        if (have_next) {
            const float4* kp = (const float4*)(ksrc + (size_t)(kb + TK + skey) * D_ + sdg);
            const float4* vp = (const float4*)(vsrc + (size_t)(kb + TK + skey) * D_ + sdg);
            #pragma unroll
            for (int q = 0; q < 4; ++q) { kr[q] = kp[q]; vr[q] = vp[q]; }
            // prefetch next bias chunk (the only cold HBM stream)
            __builtin_prefetch(bias + (size_t)l15 * S_ + (kb + TK) + half * 32, 0, 3);
        }

        // ---- S = (Q/8) @ K^T : 4 key-col tiles x 2 K-steps ----
        v8f s[4] = {};
        #pragma unroll
        for (int t = 0; t < 4; ++t)
            #pragma unroll
            for (int f = 0; f < 2; ++f) {
                v16h kfrag;
                #pragma unroll
                for (int s2 = 0; s2 < 16; ++s2)
                    kfrag[s2] = ldsK[cur][f * 32 + 8 * half + (s2 & 7) + ((s2 >> 3) << 4)][t * 16 + l15];
                s[t] = __builtin_amdgcn_wmma_f32_16x16x32_f16(
                           false, qa[f], false, kfrag, (short)0, s[t], false, false);
            }

        // ---- bias add + row max (ds_swizzle butterfly, 16-lane groups) ----
        float mnew[8];
        #pragma unroll
        for (int r = 0; r < 8; ++r) {
            const float* bp = bias + (size_t)(r + 8 * half) * S_ + kb + l15;
            float s0 = s[0][r] + bp[0];
            float s1 = s[1][r] + bp[16];
            float s2 = s[2][r] + bp[32];
            float s3 = s[3][r] + bp[48];
            s[0][r] = s0; s[1][r] = s1; s[2][r] = s2; s[3][r] = s3;
            float mx = fmaxf(fmaxf(s0, s1), fmaxf(s2, s3));
            SWZ_MAX(mx, 0x041F);   // xor 1
            SWZ_MAX(mx, 0x081F);   // xor 2
            SWZ_MAX(mx, 0x101F);   // xor 4
            SWZ_MAX(mx, 0x201F);   // xor 8
            mnew[r] = fmaxf(m_run[r], mx);
        }

        // ---- p = exp(s - m); rescale acc; stash P in A-fragment order ----
        float alpha[8];
        #pragma unroll
        for (int r = 0; r < 8; ++r) {
            alpha[r] = __expf(m_run[r] - mnew[r]);
            m_run[r] = mnew[r];
            #pragma unroll
            for (int t = 0; t < 4; ++t) {
                float p = __expf(s[t][r] - mnew[r]);
                int k  = t * 16 + l15;            // chunk-local key 0..63
                int ks = k >> 5;                  // which A-fragment
                int k5 = k & 31;
                int hsel = (k5 >> 3) & 1;         // A-layout half
                int v    = ((k5 >> 1) & 3) | (((k5 >> 4) & 1) << 2);
                int e    = (v << 1) | (k5 & 1);
                ldsP[wave][ks][hsel][r + 8 * half][e] = (_Float16)p;
            }
            #pragma unroll
            for (int t = 0; t < 4; ++t) acc[t][r] *= alpha[r];
        }
        asm volatile("s_wait_dscnt 0" ::: "memory");   // wave-local LDS RAW

        v16h pa0 = *(const v16h*)&ldsP[wave][0][half][l15][0];
        v16h pa1 = *(const v16h*)&ldsP[wave][1][half][l15][0];

        // ---- row sums of P via WMMA with all-ones B ----
        v8f rows = {};
        rows = __builtin_amdgcn_wmma_f32_16x16x32_f16(false, pa0, false, ones,
                                                      (short)0, rows, false, false);
        rows = __builtin_amdgcn_wmma_f32_16x16x32_f16(false, pa1, false, ones,
                                                      (short)0, rows, false, false);
        #pragma unroll
        for (int r = 0; r < 8; ++r) l_run[r] = l_run[r] * alpha[r] + rows[r];

        // ---- O += P @ V : 4 d-tiles x 2 K-steps, acc as C operand ----
        #pragma unroll
        for (int t = 0; t < 4; ++t)
            #pragma unroll
            for (int ks = 0; ks < 2; ++ks) {
                v16h vfrag;
                #pragma unroll
                for (int s2 = 0; s2 < 16; ++s2)
                    vfrag[s2] = ldsV[cur][ks * 32 + 8 * half + (s2 & 7) + ((s2 >> 3) << 4)][t * 16 + l15];
                acc[t] = __builtin_amdgcn_wmma_f32_16x16x32_f16(
                             false, (ks == 0) ? pa0 : pa1, false, vfrag,
                             (short)0, acc[t], false, false);
            }

        // ---- convert + store next chunk into the other buffer ----
        if (have_next) {
            const int nxt = cur ^ 1;
            #pragma unroll
            for (int q = 0; q < 4; ++q) {
                ldsK[nxt][sdg + q * 4 + 0][skey] = (_Float16)kr[q].x;
                ldsK[nxt][sdg + q * 4 + 1][skey] = (_Float16)kr[q].y;
                ldsK[nxt][sdg + q * 4 + 2][skey] = (_Float16)kr[q].z;
                ldsK[nxt][sdg + q * 4 + 3][skey] = (_Float16)kr[q].w;
            }
            v8h lo, hi;
            #pragma unroll
            for (int q = 0; q < 2; ++q) {
                lo[q*4+0]=(_Float16)vr[q].x;   lo[q*4+1]=(_Float16)vr[q].y;
                lo[q*4+2]=(_Float16)vr[q].z;   lo[q*4+3]=(_Float16)vr[q].w;
                hi[q*4+0]=(_Float16)vr[q+2].x; hi[q*4+1]=(_Float16)vr[q+2].y;
                hi[q*4+2]=(_Float16)vr[q+2].z; hi[q*4+3]=(_Float16)vr[q+2].w;
            }
            *(v8h*)&ldsV[nxt][skey][sdg]     = lo;
            *(v8h*)&ldsV[nxt][skey][sdg + 8] = hi;
        }
        __syncthreads();   // next-chunk stores visible; old-buffer reads done
    }

    // ---- finalize: O / l, coalesced fp32 store [B,H,S,D] ----
    float* op = out + (((size_t)b * H_ + h) * S_ + row0) * D_;
    #pragma unroll
    for (int r = 0; r < 8; ++r) {
        float inv = 1.0f / l_run[r];
        #pragma unroll
        for (int t = 0; t < 4; ++t)
            op[(size_t)(r + 8 * half) * D_ + t * 16 + l15] = acc[t][r] * inv;
    }
}

extern "C" void kernel_launch(void* const* d_in, const int* in_sizes, int n_in,
                              void* d_out, int out_size, void* d_ws, size_t ws_size,
                              hipStream_t stream) {
    const float* x1 = (const float*)d_in[0];
    const float* x2 = (const float*)d_in[1];
    const float* x3 = (const float*)d_in[2];
    const float* x4 = (const float*)d_in[3];
    float* outp = (float*)d_out;

    dim3 grid(B_ * H_ * (S_ / ROWS_PER_BLOCK));   // 256 blocks
    dim3 block(256);                               // 8 waves (wave32)
    hipLaunchKernelGGL(attn_bias_flash_kernel, grid, block, 0, stream,
                       x1, x2, x3, x4, outp);
}
